// SparseConv_32830730011308
// MI455X (gfx1250) — compile-verified, matrix operand order checked by reading, not run on hardware
//
#include <hip/hip_runtime.h>
#include <hip/hip_bf16.h>
#include <float.h>

// ---------------------------------------------------------------------------
// Types for CDNA5 WMMA (wave32): D(16x16 f32) = A(16x32 f16) x B(32x16 f16) + C
// ---------------------------------------------------------------------------
typedef __attribute__((ext_vector_type(16))) _Float16 v16h;
typedef __attribute__((ext_vector_type(8)))  float    v8f;
typedef __attribute__((ext_vector_type(4)))  int      v4i;

#define KNN 10
#define NBATCH 32

__device__ __forceinline__ v8f wmma32(v16h a, v16h b, v8f c) {
  // emits v_wmma_f32_16x16x32_f16
  return __builtin_amdgcn_wmma_f32_16x16x32_f16(false, a, false, b, (short)0, c,
                                                false, false);
}

// K-offset of fragment element pair i for this lane-half (ISA 7.12.2, 16-bit A 16x32)
__device__ __host__ __forceinline__ int frag_k(int i, int half) {
  return (i < 4) ? (half * 8 + 2 * i) : (16 + half * 8 + 2 * (i - 4));
}

// Load A fragment (16x32 f16) from LDS, row-major with given stride (f16 units).
__device__ __forceinline__ v16h load_a_lds(const _Float16* base, int stride,
                                           int row0, int kc, int lane) {
  const int r = row0 + (lane & 15);
  const int half = lane >> 4;
  const _Float16* p = base + r * stride + kc * 32;
  v16h a;
#pragma unroll
  for (int i = 0; i < 8; ++i) {
    int k = frag_k(i, half);
    a[2 * i]     = p[k];
    a[2 * i + 1] = p[k + 1];
  }
  return a;
}

// Load a pre-packed B fragment: one aligned 32B vector per lane.
__device__ __forceinline__ v16h load_b_packed(const _Float16* __restrict__ P,
                                              int NT, int kc, int nt, int lane) {
  return *(const v16h*)(P + (size_t)(((kc * NT) + nt) * 32 + lane) * 16);
}

// ---------------------------------------------------------------------------
// One-time weight pre-pack: fragment-order f16, zero-padded.
//   out[((kc*NT+nt)*32 + lane)*16 + 2*i + s] = W[kc*32 + frag_k(i,half) + s][nt*16 + n]
// ---------------------------------------------------------------------------
__global__ void pack_weights_kernel(const float* __restrict__ W, int Krows,
                                    int Ncols, int KC, int NT,
                                    _Float16* __restrict__ out) {
  int t = blockIdx.x * blockDim.x + threadIdx.x;
  int total = KC * NT * 32 * 16;
  if (t >= total) return;
  int e = t & 15;
  int lane = (t >> 4) & 31;
  int frag = t >> 9;            // kc*NT + nt
  int nt = frag % NT;
  int kc = frag / NT;
  int i = e >> 1, s = e & 1;
  int half = lane >> 4;
  int k = kc * 32 + frag_k(i, half) + s;
  int n = nt * 16 + (lane & 15);
  out[t] = (k < Krows && n < Ncols) ? (_Float16)W[(size_t)k * Ncols + n]
                                    : (_Float16)0.f;
}

// ---------------------------------------------------------------------------
// Brute-force kNN (K=10, D=3), thread per (b,m); insertion top-k, ties -> lower idx
// ---------------------------------------------------------------------------
__global__ void knn_kernel(const float* __restrict__ coords, int M, int total,
                           int* __restrict__ idx_out) {
  int t = blockIdx.x * blockDim.x + threadIdx.x;
  if (t >= total) return;
  int b = t / M, m = t % M;
  const float* cb = coords + (size_t)b * M * 3;
  float x = cb[m * 3 + 0], y = cb[m * 3 + 1], z = cb[m * 3 + 2];
  float dbest[KNN];
  int   ibest[KNN];
#pragma unroll
  for (int i = 0; i < KNN; ++i) { dbest[i] = FLT_MAX; ibest[i] = 0; }
  for (int j = 0; j < M; ++j) {
    float dx = cb[j * 3 + 0] - x;
    float dy = cb[j * 3 + 1] - y;
    float dz = cb[j * 3 + 2] - z;
    float d = dx * dx + dy * dy + dz * dz;
    if (d < dbest[KNN - 1]) {
      int pos = KNN - 1;
      while (pos > 0 && d < dbest[pos - 1]) {
        dbest[pos] = dbest[pos - 1]; ibest[pos] = ibest[pos - 1]; --pos;
      }
      dbest[pos] = d; ibest[pos] = j;
    }
  }
#pragma unroll
  for (int k = 0; k < KNN; ++k) idx_out[(size_t)t * KNN + k] = ibest[k];
}

// ---------------------------------------------------------------------------
// Sparse conv: one wave32 block handles 16 points of one batch.
// Each point owns 16 A-rows (K=10 neighbors + 6 zero rows) so that in GEMM1 one
// row-tile == one point: no integer division in the epilogue.
//   GEMM1: g[16pts*16, F_IN] @ W1 + b1, relu  -> h (LDS, flattened per point)
//   GEMM2: h[16, K*OUT] @ W2 + b2, relu       -> out
// ---------------------------------------------------------------------------
template <int F_IN, int OUT>
__global__ __launch_bounds__(32)
void sparse_conv_kernel(const float* __restrict__ feat_in,  // [B,M,F_IN]
                        const int*   __restrict__ knn_idx,  // [B,M,KNN]
                        const _Float16* __restrict__ pW1,   // packed frags
                        const float* __restrict__ b1,
                        const _Float16* __restrict__ pW2,   // packed frags
                        const float* __restrict__ b2,
                        float* __restrict__ feat_out,       // [B,M,OUT]
                        int M, float scale) {
  constexpr int KPAD1 = ((F_IN + 31) / 32) * 32;
  constexpr int NT    = (OUT + 15) / 16;
  constexpr int KW    = KNN * OUT;
  constexpr int KPAD2 = ((KW + 31) / 32) * 32;
  constexpr int KC1   = KPAD1 / 32;
  constexpr int KC2   = KPAD2 / 32;
  constexpr int ROWS  = 16 * 16;          // 16 points x 16 padded neighbor rows

  __shared__ __align__(16) _Float16 gA[ROWS * KPAD1];
  __shared__ __align__(16) _Float16 hA[16 * KPAD2];
  __shared__ int nidx[16 * KNN];

  const int lane = threadIdx.x;
  const int half = lane >> 4;
  const int tiles_per_b = (M + 15) / 16;
  const int b  = blockIdx.x / tiles_per_b;
  const int m0 = (blockIdx.x % tiles_per_b) * 16;

  // phase 0: zero LDS (b128 stores), stage neighbor indices
  {
    v4i z = {0, 0, 0, 0};
    v4i* zg = (v4i*)gA;
    for (int i = lane; i < ROWS * KPAD1 / 8; i += 32) zg[i] = z;
    v4i* zh = (v4i*)hA;
    for (int i = lane; i < 16 * KPAD2 / 8; i += 32) zh[i] = z;
  }
  for (int r = lane; r < 16 * KNN; r += 32) {
    int p = r / KNN, kk = r % KNN, m = m0 + p;
    nidx[r] = (m < M) ? knn_idx[((size_t)b * M + m) * KNN + kk] : -1;
  }
  __syncthreads();

  // phase 1: gather neighbor features into gA rows p*16+kk (f32 -> f16, scaled)
  for (int e = lane; e < 16 * KNN * F_IN; e += 32) {
    int p = e / (KNN * F_IN);
    int rem = e - p * (KNN * F_IN);
    int kk = rem / F_IN;
    int f = rem - kk * F_IN;
    int j = nidx[p * KNN + kk];
    if (j >= 0)
      gA[(p * 16 + kk) * KPAD1 + f] =
          (_Float16)(feat_in[((size_t)b * M + j) * F_IN + f] * scale);
  }
  __syncthreads();

  // phase 2: GEMM1 — one row tile per point; bias+relu; scatter into hA.
  // Store guard decomposition (kk = j + 8*half < 10):
  //   j=0,1 -> valid for both halves; j=2..7 -> valid only for half==0.
#pragma unroll 1
  for (int nt = 0; nt < NT; ++nt) {
    v16h bfr[KC1];
#pragma unroll
    for (int kc = 0; kc < KC1; ++kc)
      bfr[kc] = load_b_packed(pW1, NT, kc, nt, lane);
    const int c = nt * 16 + (lane & 15);
    const float bias = (c < OUT) ? b1[c] : 0.f;
#pragma unroll 2
    for (int pt = 0; pt < 16; ++pt) {
      v8f acc = {0.f, 0.f, 0.f, 0.f, 0.f, 0.f, 0.f, 0.f};
#pragma unroll
      for (int kc = 0; kc < KC1; ++kc)
        acc = wmma32(load_a_lds(gA, KPAD1, pt * 16, kc, lane), bfr[kc], acc);
      if (c < OUT) {
        // rows kk = 8*half + {0,1}: unconditional, constant offsets off one base
        _Float16* hb = &hA[pt * KPAD2 + (8 * half) * OUT + c];
        float v0 = acc[0] + bias;
        float v1 = acc[1] + bias;
        hb[0]   = (_Float16)(v0 > 0.f ? v0 : 0.f);
        hb[OUT] = (_Float16)(v1 > 0.f ? v1 : 0.f);
        if (half == 0) {                   // single exec region, 6 const-offset stores
          _Float16* hc = &hA[pt * KPAD2 + c];
#pragma unroll
          for (int j = 2; j < 8; ++j) {
            float v = acc[j] + bias;
            hc[j * OUT] = (_Float16)(v > 0.f ? v : 0.f);
          }
        }
      }
    }
  }
  __syncthreads();

  // phase 3: GEMM2 — single 16-row tile, dual accumulators to pipeline
#pragma unroll 1
  for (int nt = 0; nt < NT; ++nt) {
    v8f acc0 = {0.f, 0.f, 0.f, 0.f, 0.f, 0.f, 0.f, 0.f};
    v8f acc1 = {0.f, 0.f, 0.f, 0.f, 0.f, 0.f, 0.f, 0.f};
#pragma unroll 1
    for (int kc = 0; kc + 1 < KC2; kc += 2) {
      acc0 = wmma32(load_a_lds(hA, KPAD2, 0, kc, lane),
                    load_b_packed(pW2, NT, kc, nt, lane), acc0);
      acc1 = wmma32(load_a_lds(hA, KPAD2, 0, kc + 1, lane),
                    load_b_packed(pW2, NT, kc + 1, nt, lane), acc1);
    }
    if (KC2 & 1) {
      acc0 = wmma32(load_a_lds(hA, KPAD2, 0, KC2 - 1, lane),
                    load_b_packed(pW2, NT, KC2 - 1, nt, lane), acc0);
    }
    const int c = nt * 16 + (lane & 15);
    if (c < OUT) {
      const float bias = b2[c];
      if (m0 + 16 <= M) {
        // full tile (common case): scalar branch, unconditional const-offset stores
        float* ob = feat_out + ((size_t)b * M + m0 + 8 * half) * OUT + c;
#pragma unroll
        for (int j = 0; j < 8; ++j) {
          float v = acc0[j] + acc1[j] + bias;
          ob[(size_t)j * OUT] = v > 0.f ? v : 0.f;
        }
      } else {
#pragma unroll
        for (int j = 0; j < 8; ++j) {
          int m = m0 + j + 8 * half;
          if (m < M) {
            float v = acc0[j] + acc1[j] + bias;
            feat_out[((size_t)b * M + m) * OUT + c] = v > 0.f ? v : 0.f;
          }
        }
      }
    }
  }
}

// ---------------------------------------------------------------------------
// Pooling support
// ---------------------------------------------------------------------------
__global__ void rowmax_kernel(const float* __restrict__ feat, int M, int C,
                              int total, float* __restrict__ out) {
  int t = blockIdx.x * blockDim.x + threadIdx.x;
  if (t >= total) return;
  const float* p = feat + (size_t)t * C;
  float m = p[0];
  for (int c = 1; c < C; ++c) m = p[c] > m ? p[c] : m;
  out[t] = m;
}

// Per-batch bitonic sort, descending by value, ties -> ascending index
// (matches jax.lax.top_k ordering). First n_out indices written out.
template <int SZ>
__global__ void topk_kernel(const float* __restrict__ rowmax, int M, int n_out,
                            int* __restrict__ pidx) {
  __shared__ float v[SZ];
  __shared__ int   id[SZ];
  const int b = blockIdx.x;
  for (int i = threadIdx.x; i < SZ; i += blockDim.x) {
    v[i]  = (i < M) ? rowmax[(size_t)b * M + i] : -FLT_MAX;
    id[i] = i;
  }
  __syncthreads();
  for (int k = 2; k <= SZ; k <<= 1) {
    for (int j = k >> 1; j > 0; j >>= 1) {
      for (int i = threadIdx.x; i < SZ; i += blockDim.x) {
        int ixj = i ^ j;
        if (ixj > i) {
          bool before = (v[i] > v[ixj]) || (v[i] == v[ixj] && id[i] < id[ixj]);
          bool desc_seg = ((i & k) == 0);
          if (desc_seg ? !before : before) {
            float tv = v[i]; v[i] = v[ixj]; v[ixj] = tv;
            int ti = id[i]; id[i] = id[ixj]; id[ixj] = ti;
          }
        }
      }
      __syncthreads();
    }
  }
  for (int i = threadIdx.x; i < n_out; i += blockDim.x)
    pidx[(size_t)b * n_out + i] = id[i];
}

__global__ void pool_gather_kernel(const float* __restrict__ feat,
                                   const float* __restrict__ sg,
                                   const float* __restrict__ sl,
                                   const int* __restrict__ pidx,
                                   int M, int C, int n_out, int total,
                                   float* __restrict__ feat_o,
                                   float* __restrict__ sg_o,
                                   float* __restrict__ sl_o) {
  int t = blockIdx.x * blockDim.x + threadIdx.x;
  if (t >= total) return;
  int b = t / n_out;
  int src = pidx[t];
  const float* fs = feat + ((size_t)b * M + src) * C;
  float* fd = feat_o + (size_t)t * C;
  for (int c = 0; c < C; ++c) fd[c] = fs[c];
  const float* gs = sg + ((size_t)b * M + src) * 3;
  float* gd = sg_o + (size_t)t * 3;
  gd[0] = gs[0]; gd[1] = gs[1]; gd[2] = gs[2];
  const float* ls = sl + ((size_t)b * M + src) * 2;
  float* ld = sl_o + (size_t)t * 2;
  ld[0] = ls[0]; ld[1] = ls[1];
}

// ---------------------------------------------------------------------------
// FC head (tiny; plain f32)
// ---------------------------------------------------------------------------
__global__ void fc1_kernel(const float* __restrict__ allf,  // [B,500*80] flat
                           const float* __restrict__ sg,    // [B,500*3]
                           const float* __restrict__ sl,    // [B,500*2]
                           const float* __restrict__ W,     // [42500,100]
                           const float* __restrict__ bias,
                           float* __restrict__ out) {       // [B,100]
  int t = blockIdx.x * blockDim.x + threadIdx.x;
  if (t >= NBATCH * 100) return;
  int b = t / 100, o = t % 100;
  float acc = bias[o];
  const float* fa = allf + (size_t)b * 40000;
  for (int k = 0; k < 40000; ++k) acc += fa[k] * W[(size_t)k * 100 + o];
  const float* fg = sg + (size_t)b * 1500;
  for (int k = 0; k < 1500; ++k) acc += fg[k] * W[(size_t)(40000 + k) * 100 + o];
  const float* fl = sl + (size_t)b * 1000;
  for (int k = 0; k < 1000; ++k) acc += fl[k] * W[(size_t)(41500 + k) * 100 + o];
  out[t] = acc > 0.f ? acc : 0.f;
}

__global__ void fc_kernel(const float* __restrict__ in, const float* __restrict__ W,
                          const float* __restrict__ bias, float* __restrict__ out,
                          int DIN, int DOUT, int relu) {
  int t = blockIdx.x * blockDim.x + threadIdx.x;
  if (t >= NBATCH * DOUT) return;
  int b = t / DOUT, o = t % DOUT;
  float acc = bias[o];
  const float* p = in + (size_t)b * DIN;
  for (int k = 0; k < DIN; ++k) acc += p[k] * W[(size_t)k * DOUT + o];
  out[t] = (relu && acc < 0.f) ? 0.f : acc;
}

// ---------------------------------------------------------------------------
// Host launcher
// ---------------------------------------------------------------------------
extern "C" void kernel_launch(void* const* d_in, const int* in_sizes, int n_in,
                              void* d_out, int out_size, void* d_ws, size_t ws_size,
                              hipStream_t stream) {
  (void)in_sizes; (void)n_in; (void)out_size; (void)ws_size;
  const int B = NBATCH, N = 2048, P1 = 1000, P2 = 500;
  const int FIN[5] = {13, 15, 30, 45, 60};
  const int OUTC[5] = {15, 30, 45, 60, 80};

  const float* space  = (const float*)d_in[0];      // [B,2048,3]
  const float* slocal = (const float*)d_in[1];      // [B,2048,2]
  const float* allf   = (const float*)d_in[2];      // [B,2048,13]
  // d_in[3] = num_entries (unused by reference math)
  const float* cW1[5]; const float* cb1[5]; const float* cW2[5]; const float* cb2[5];
  for (int i = 0; i < 5; ++i) {
    cW1[i] = (const float*)d_in[4 + 4 * i];
    cb1[i] = (const float*)d_in[5 + 4 * i];
    cW2[i] = (const float*)d_in[6 + 4 * i];
    cb2[i] = (const float*)d_in[7 + 4 * i];
  }
  const float* fW[3]; const float* fb[3];
  for (int i = 0; i < 3; ++i) {
    fW[i] = (const float*)d_in[24 + 2 * i];
    fb[i] = (const float*)d_in[25 + 2 * i];
  }
  float* out = (float*)d_out;

  // bump allocator over workspace
  size_t off = 0;
  auto alloc = [&](size_t nbytes) -> void* {
    void* p = (char*)d_ws + off;
    off += (nbytes + 255) & ~(size_t)255;
    return p;
  };
  int*   idx0  = (int*)  alloc((size_t)B * N  * KNN * 4);
  float* f1    = (float*)alloc((size_t)B * N  * 15 * 4);
  float* f2    = (float*)alloc((size_t)B * N  * 30 * 4);
  float* f3    = (float*)alloc((size_t)B * N  * 45 * 4);
  float* rmax  = (float*)alloc((size_t)B * N  * 4);
  int*   pidx1 = (int*)  alloc((size_t)B * P1 * 4);
  float* f3p   = (float*)alloc((size_t)B * P1 * 45 * 4);
  float* sg1   = (float*)alloc((size_t)B * P1 * 3 * 4);
  float* sl1   = (float*)alloc((size_t)B * P1 * 2 * 4);
  int*   idx1  = (int*)  alloc((size_t)B * P1 * KNN * 4);
  float* f4    = (float*)alloc((size_t)B * P1 * 60 * 4);
  int*   pidx2 = (int*)  alloc((size_t)B * P2 * 4);
  float* f4p   = (float*)alloc((size_t)B * P2 * 60 * 4);
  float* sg2   = (float*)alloc((size_t)B * P2 * 3 * 4);
  float* sl2   = (float*)alloc((size_t)B * P2 * 2 * 4);
  int*   idx2  = (int*)  alloc((size_t)B * P2 * KNN * 4);
  float* f5    = (float*)alloc((size_t)B * P2 * 80 * 4);
  float* fcb1  = (float*)alloc((size_t)B * 100 * 4);
  float* fcb2  = (float*)alloc((size_t)B * 30 * 4);

  // packed weight fragments (f16, fragment lane order)
  _Float16* pW1[5];
  _Float16* pW2[5];
  int KC1v[5], KC2v[5], NTv[5];
  for (int i = 0; i < 5; ++i) {
    KC1v[i] = (FIN[i] + 31) / 32;
    NTv[i]  = (OUTC[i] + 15) / 16;
    KC2v[i] = (KNN * OUTC[i] + 31) / 32;
    pW1[i] = (_Float16*)alloc((size_t)KC1v[i] * NTv[i] * 32 * 16 * 2);
    pW2[i] = (_Float16*)alloc((size_t)KC2v[i] * NTv[i] * 32 * 16 * 2);
  }
  for (int i = 0; i < 5; ++i) {
    int t1 = KC1v[i] * NTv[i] * 512;
    pack_weights_kernel<<<(t1 + 255) / 256, 256, 0, stream>>>(
        cW1[i], FIN[i], OUTC[i], KC1v[i], NTv[i], pW1[i]);
    int t2 = KC2v[i] * NTv[i] * 512;
    pack_weights_kernel<<<(t2 + 255) / 256, 256, 0, stream>>>(
        cW2[i], KNN * OUTC[i], OUTC[i], KC2v[i], NTv[i], pW2[i]);
  }

  // stage 1: kNN @2048 shared by convs 0-2
  {
    int total = B * N;
    knn_kernel<<<(total + 255) / 256, 256, 0, stream>>>(space, N, total, idx0);
  }
  sparse_conv_kernel<13, 15><<<B * (N / 16), 32, 0, stream>>>(
      allf, idx0, pW1[0], cb1[0], pW2[0], cb2[0], f1, N, 0.001f);
  sparse_conv_kernel<15, 30><<<B * (N / 16), 32, 0, stream>>>(
      f1, idx0, pW1[1], cb1[1], pW2[1], cb2[1], f2, N, 1.0f);
  sparse_conv_kernel<30, 45><<<B * (N / 16), 32, 0, stream>>>(
      f2, idx0, pW1[2], cb1[2], pW2[2], cb2[2], f3, N, 1.0f);

  // pool 2048 -> 1000
  rowmax_kernel<<<(B * N + 255) / 256, 256, 0, stream>>>(f3, N, 45, B * N, rmax);
  topk_kernel<2048><<<B, 1024, 0, stream>>>(rmax, N, P1, pidx1);
  pool_gather_kernel<<<(B * P1 + 255) / 256, 256, 0, stream>>>(
      f3, space, slocal, pidx1, N, 45, P1, B * P1, f3p, sg1, sl1);

  // conv3 @1000
  {
    int total = B * P1;
    knn_kernel<<<(total + 255) / 256, 256, 0, stream>>>(sg1, P1, total, idx1);
  }
  sparse_conv_kernel<45, 60><<<B * ((P1 + 15) / 16), 32, 0, stream>>>(
      f3p, idx1, pW1[3], cb1[3], pW2[3], cb2[3], f4, P1, 1.0f);

  // pool 1000 -> 500
  rowmax_kernel<<<(B * P1 + 255) / 256, 256, 0, stream>>>(f4, P1, 60, B * P1, rmax);
  topk_kernel<1024><<<B, 512, 0, stream>>>(rmax, P1, P2, pidx2);
  pool_gather_kernel<<<(B * P2 + 255) / 256, 256, 0, stream>>>(
      f4, sg1, sl1, pidx2, P1, 60, P2, B * P2, f4p, sg2, sl2);

  // conv4 @500
  {
    int total = B * P2;
    knn_kernel<<<(total + 255) / 256, 256, 0, stream>>>(sg2, P2, total, idx2);
  }
  sparse_conv_kernel<60, 80><<<B * ((P2 + 15) / 16), 32, 0, stream>>>(
      f4p, idx2, pW1[4], cb1[4], pW2[4], cb2[4], f5, P2, 1.0f);

  // FC head
  fc1_kernel<<<(B * 100 + 255) / 256, 256, 0, stream>>>(f5, sg2, sl2, fW[0], fb[0], fcb1);
  fc_kernel<<<(B * 30 + 255) / 256, 256, 0, stream>>>(fcb1, fW[1], fb[1], fcb2, 100, 30, 1);
  fc_kernel<<<(B * 6 + 255) / 256, 256, 0, stream>>>(fcb2, fW[2], fb[2], out, 30, 6, 0);
}
